// MOE_7456063225884
// MI455X (gfx1250) — matrix operand order repaired and checked by reading, not compiled, not run on hardware
//
#include <hip/hip_runtime.h>
#include <hip/hip_bf16.h>
#include <math.h>
#include <stdint.h>

#define T_TOKENS 8192
#define D_IN     2048
#define D_INTER  1024
#define NEXP     32
#define CAP      512
#define ASTRIDE  40   // padded LDS row stride in halves (80B: 16B-aligned, 2-way banks)

typedef __attribute__((ext_vector_type(16))) _Float16 v16h;
typedef __attribute__((ext_vector_type(8)))  _Float16 v8h;
typedef __attribute__((ext_vector_type(4)))  _Float16 v4h;
typedef __attribute__((ext_vector_type(8)))  float    v8f;
typedef unsigned int u32x4 __attribute__((ext_vector_type(4)));
typedef int          i32x4 __attribute__((ext_vector_type(4)));
typedef int          i32x8 __attribute__((ext_vector_type(8)));

#if __has_builtin(__builtin_amdgcn_tensor_store_from_lds) && \
    __has_builtin(__builtin_amdgcn_s_wait_tensorcnt)
#define HAVE_TDM 1
#endif

// ---------------------------------------------------------------------------
// Async global->LDS copy (CDNA5, tracked by ASYNCcnt). LDS address is the low
// 32 bits of the generic pointer (ISA flat->LDS mapping).
// ---------------------------------------------------------------------------
__device__ __forceinline__ void async_copy_b64(unsigned lds_off, const void* gptr) {
    asm volatile("global_load_async_to_lds_b64 %0, %1, off"
                 :: "v"(lds_off), "v"(gptr) : "memory");
}
__device__ __forceinline__ void wait_async_le1() {
    asm volatile("s_wait_asynccnt 0x1" ::: "memory");
}
__device__ __forceinline__ void wait_async_le0() {
    asm volatile("s_wait_asynccnt 0x0" ::: "memory");
}

#ifdef HAVE_TDM
// 2D TDM store: LDS (packed rows) -> global (row pitch in elements).
// Descriptor per ISA 8.3/8.4: group0 {count=1, lds_addr, global_addr, type=2};
// group1 {data_size=2B, tile 256x32, tensor dims, dim0 stride}.
__device__ __forceinline__ void tdm_store_tile_f16(
    unsigned lds_off, const void* gaddr, int tile_w, int tile_h, int pitch)
{
    const uint64_t ga = (uint64_t)(uintptr_t)gaddr;
    u32x4 g0;
    g0.x = 1u;                                   // count=1, user mode
    g0.y = lds_off;                              // lds_addr
    g0.z = (unsigned)ga;                         // global_addr[31:0]
    g0.w = (unsigned)(ga >> 32) | (2u << 30);    // global_addr[56:32] | type=2
    i32x8 g1;
    g1[0] = 1 << 16;                             // data_size=1 (2 bytes)
    g1[1] = tile_w << 16;                        // tensor_dim0[15:0]
    g1[2] = tile_h << 16;                        // tensor_dim1[15:0] (dim0 hi=0)
    g1[3] = tile_w << 16;                        // tile_dim0 (dim1 hi=0)
    g1[4] = tile_h;                              // tile_dim1 (tile_dim2=0)
    g1[5] = pitch;                               // tensor_dim0_stride[31:0]
    g1[6] = 0;
    g1[7] = 0;
    const i32x4 z4 = {};
#if __has_include(<hip/amd_detail/amd_gfx1250_TDM.h>)
    const i32x8 z8 = {};
    __builtin_amdgcn_tensor_store_from_lds(g0, g1, z4, z4, z8, 0);
#else
    __builtin_amdgcn_tensor_store_from_lds(g0, g1, z4, z4, 0);
#endif
}
#endif

// ---------------------------------------------------------------------------
// Kernel 0: per-expert ordered compaction (reproduces cumsum position logic).
// ---------------------------------------------------------------------------
__global__ __launch_bounds__(32) void moe_dispatch_kernel(
    const int* __restrict__ idx, int* __restrict__ counts, int* __restrict__ toklist)
{
    const int e    = blockIdx.x;
    const int lane = threadIdx.x;
    int base = 0;
    for (int i = 0; i < T_TOKENS; i += 32) {
        const int t = i + lane;
        const bool m = (idx[t] == e);
        const unsigned mask = (unsigned)__ballot(m);
        if (m) {
            const int pos = base + __popc(mask & ((1u << lane) - 1u));
            if (pos < CAP) toklist[e * CAP + pos] = t;
        }
        base += __popc(mask);
    }
    if (lane == 0) counts[e] = (base < CAP) ? base : CAP;
}

// ---------------------------------------------------------------------------
// WMMA fragment helpers (ISA 7.12.2 layouts, wave32).
// ---------------------------------------------------------------------------
__device__ __forceinline__ v16h load_a_frag(const _Float16* As, int wm, int lane) {
    const _Float16* ap = As + (wm * 16 + (lane & 15)) * ASTRIDE;
    const int kb = (lane < 16) ? 0 : 8;
    const v8h lo = *(const v8h*)(ap + kb);
    const v8h hi = *(const v8h*)(ap + kb + 16);
    v16h a;
#pragma unroll
    for (int i = 0; i < 8; ++i) { a[i] = lo[i]; a[i + 8] = hi[i]; }
    return a;
}

__device__ __forceinline__ v16h load_b_frag(const _Float16* Bs, int ncol, int lane) {
    // Bs layout: [n][k], padded row stride ASTRIDE halves
    const _Float16* bp = Bs + (ncol + (lane & 15)) * ASTRIDE + ((lane < 16) ? 0 : 8);
    const v8h lo = *(const v8h*)(bp);
    const v8h hi = *(const v8h*)(bp + 16);
    v16h b;
#pragma unroll
    for (int i = 0; i < 8; ++i) { b[i] = lo[i]; b[i + 8] = hi[i]; }
    return b;
}

// ---------------------------------------------------------------------------
// Kernel 1: H = silu(X·W1 + b1) * (X·W3 + b3), f16 H to ws.
// Grid: (m-tile, n-tile, expert): consecutive blocks share weight slices (L2).
// Pipelined global loads; H tile staged in LDS and written by one TDM store.
// ---------------------------------------------------------------------------
__global__ __launch_bounds__(256) void moe_ffn1_kernel(
    const float* __restrict__ x,
    const float* __restrict__ W1, const float* __restrict__ b1,
    const float* __restrict__ W3, const float* __restrict__ b3,
    const int* __restrict__ counts, const int* __restrict__ toklist,
    _Float16* __restrict__ H)
{
    const int m0 = blockIdx.x * 32;
    const int n0 = blockIdx.y * 256;
    const int e  = blockIdx.z;
    const int count = counts[e];
    if (m0 >= count) return;   // uniform exit before any barrier

    __shared__ _Float16 As[32 * ASTRIDE];
    __shared__ _Float16 Bs1[256 * ASTRIDE];
    __shared__ _Float16 Bs3[256 * ASTRIDE];
#ifdef HAVE_TDM
    __shared__ _Float16 Hs[32 * 256];   // packed H tile for TDM store
#endif

    const int tid  = threadIdx.x;
    const int lane = tid & 31;
    const int wave = tid >> 5;
    const int wm   = wave & 1;
    const int wn   = wave >> 1;

    // A gather: one float4 chunk per thread; row = tid/8, k = (tid%8)*4.
    // Clamped rows only affect H rows that are never consumed.
    const int arow = tid >> 3;
    const int akk  = (tid & 7) * 4;
    const int grc  = (m0 + arow < count) ? (m0 + arow) : (count - 1);
    const float* aptr = x + (size_t)toklist[e * CAP + grc] * D_IN + akk;

    const size_t wbase = (size_t)e * D_IN * D_INTER;
    const float* W1e = W1 + wbase;
    const float* W3e = W3 + wbase;

    v8f acc1[4], acc3[4];
    const v8f vz = {};
#pragma unroll
    for (int s = 0; s < 4; ++s) { acc1[s] = vz; acc3[s] = vz; }

    float4 rfa;
    float r1[32], r3[32];
    auto issue_loads = [&](int k0) {
        rfa = *(const float4*)(aptr + k0);
        const float* p1 = W1e + (size_t)k0 * D_INTER + (n0 + tid);
        const float* p3 = W3e + (size_t)k0 * D_INTER + (n0 + tid);
#pragma unroll
        for (int i = 0; i < 32; ++i) {
            r1[i] = p1[(size_t)i * D_INTER];
            r3[i] = p3[(size_t)i * D_INTER];
        }
        __builtin_prefetch(p1 + (size_t)32 * D_INTER, 0, 1);
        __builtin_prefetch(p3 + (size_t)32 * D_INTER, 0, 1);
    };

    issue_loads(0);

    for (int k0 = 0; k0 < D_IN; k0 += 32) {
        // Convert pending registers, stage tile to LDS (load wait lands here).
        v4h av;
        av[0] = (_Float16)rfa.x; av[1] = (_Float16)rfa.y;
        av[2] = (_Float16)rfa.z; av[3] = (_Float16)rfa.w;
        *(v4h*)&As[arow * ASTRIDE + akk] = av;
        _Float16 t1[32], t3[32];
#pragma unroll
        for (int i = 0; i < 32; ++i) { t1[i] = (_Float16)r1[i]; t3[i] = (_Float16)r3[i]; }
        v8h* d1 = (v8h*)&Bs1[tid * ASTRIDE];
        v8h* d3 = (v8h*)&Bs3[tid * ASTRIDE];
#pragma unroll
        for (int q = 0; q < 4; ++q) {
            d1[q] = *(const v8h*)&t1[q * 8];
            d3[q] = *(const v8h*)&t3[q * 8];
        }
        __syncthreads();

        if (k0 + 32 < D_IN) issue_loads(k0 + 32);  // overlap with WMMAs below

        const v16h a = load_a_frag(As, wm, lane);
        v16h cb1 = load_b_frag(Bs1, wn * 64, lane);
        v16h cb3 = load_b_frag(Bs3, wn * 64, lane);
#pragma unroll
        for (int s = 0; s < 4; ++s) {
            v16h nb1, nb3;
            if (s < 3) {
                nb1 = load_b_frag(Bs1, wn * 64 + (s + 1) * 16, lane);
                nb3 = load_b_frag(Bs3, wn * 64 + (s + 1) * 16, lane);
            }
            acc1[s] = __builtin_amdgcn_wmma_f32_16x16x32_f16(
                false, a, false, cb1, (short)0, acc1[s], false, false);
            acc3[s] = __builtin_amdgcn_wmma_f32_16x16x32_f16(
                false, a, false, cb3, (short)0, acc3[s], false, false);
            if (s < 3) { cb1 = nb1; cb3 = nb3; }
        }
        __syncthreads();
    }

    // Epilogue: SwiGLU.
    const int mloc = wm * 16 + ((lane >> 4) << 3);
    const int nloc = wn * 64 + (lane & 15);
#pragma unroll
    for (int s = 0; s < 4; ++s) {
        const int nl = nloc + s * 16;
        const float bias1 = b1[(size_t)e * D_INTER + n0 + nl];
        const float bias3 = b3[(size_t)e * D_INTER + n0 + nl];
#pragma unroll
        for (int i = 0; i < 8; ++i) {
            const float a1 = acc1[s][i] + bias1;
            const float a3 = acc3[s][i] + bias3;
            const float hv = (a1 / (1.0f + __expf(-a1))) * a3;
#ifdef HAVE_TDM
            Hs[(mloc + i) * 256 + nl] = (_Float16)hv;
#else
            H[((size_t)e * CAP + m0 + mloc + i) * D_INTER + n0 + nl] = (_Float16)hv;
#endif
        }
    }
#ifdef HAVE_TDM
    __syncthreads();                 // H tile complete in LDS
    if (wave == 0) {                 // one TDM store per block (EXEC-independent)
        tdm_store_tile_f16((unsigned)(uintptr_t)&Hs[0],
                           H + ((size_t)e * CAP + m0) * D_INTER + n0,
                           /*tile_w=*/256, /*tile_h=*/32, /*pitch=*/D_INTER);
        __builtin_amdgcn_s_wait_tensorcnt(0);
    }
#endif
}

// ---------------------------------------------------------------------------
// Kernel 2: Out = H·W2 + b2, scattered back to token rows.
// A (H) is f16 in global: moved via async global->LDS (ASYNCcnt),
// double-buffered in LDS; B staged through registers with f32->f16 convert.
// ---------------------------------------------------------------------------
__global__ __launch_bounds__(256) void moe_ffn2_kernel(
    const _Float16* __restrict__ H,
    const float* __restrict__ W2, const float* __restrict__ b2,
    const int* __restrict__ counts, const int* __restrict__ toklist,
    float* __restrict__ out)
{
    const int m0 = blockIdx.x * 32;
    const int n0 = blockIdx.y * 256;
    const int e  = blockIdx.z;
    const int count = counts[e];
    if (m0 >= count) return;

    __shared__ _Float16 As2[2][32 * ASTRIDE];
    __shared__ _Float16 Bs[256 * ASTRIDE];

    const int tid  = threadIdx.x;
    const int lane = tid & 31;
    const int wave = tid >> 5;
    const int wm   = wave & 1;
    const int wn   = wave >> 1;

    // A: one b64 (4 halves) async chunk per thread; row = tid/8, k = (tid%8)*4.
    const int arow = tid >> 3;
    const int akk  = (tid & 7) * 4;
    const int rowc = (m0 + arow < count) ? (m0 + arow) : (count - 1);
    const _Float16* hrow = H + ((size_t)e * CAP + rowc) * D_INTER + akk;
    const unsigned lds_a[2] = {
        (unsigned)(uintptr_t)&As2[0][arow * ASTRIDE + akk],
        (unsigned)(uintptr_t)&As2[1][arow * ASTRIDE + akk]
    };

    const float* W2e = W2 + (size_t)e * D_INTER * D_IN;

    v8f acc[4];
    const v8f vz = {};
#pragma unroll
    for (int s = 0; s < 4; ++s) acc[s] = vz;

    float r2[32];
    auto issue_b = [&](int k0) {
        const float* p2 = W2e + (size_t)k0 * D_IN + (n0 + tid);
#pragma unroll
        for (int i = 0; i < 32; ++i)
            r2[i] = p2[(size_t)i * D_IN];
        __builtin_prefetch(p2 + (size_t)32 * D_IN, 0, 1);
    };

    async_copy_b64(lds_a[0], hrow);
    issue_b(0);

    int buf = 0;
    for (int k0 = 0; k0 < D_INTER; k0 += 32, buf ^= 1) {
        // Stage B tile (loadcnt wait lands in the converts).
        _Float16 t2[32];
#pragma unroll
        for (int i = 0; i < 32; ++i) t2[i] = (_Float16)r2[i];
        v8h* d2 = (v8h*)&Bs[tid * ASTRIDE];
#pragma unroll
        for (int q = 0; q < 4; ++q)
            d2[q] = *(const v8h*)&t2[q * 8];

        if (k0 + 32 < D_INTER) {
            async_copy_b64(lds_a[buf ^ 1], hrow + k0 + 32);  // next A tile
            issue_b(k0 + 32);                                // next B tile
            wait_async_le1();   // async ops retire in order: tile k0 has landed
        } else {
            wait_async_le0();
        }
        __syncthreads();

        const v16h a = load_a_frag(As2[buf], wm, lane);
        v16h cb = load_b_frag(Bs, wn * 64, lane);
#pragma unroll
        for (int s = 0; s < 4; ++s) {
            v16h nb;
            if (s < 3) nb = load_b_frag(Bs, wn * 64 + (s + 1) * 16, lane);
            acc[s] = __builtin_amdgcn_wmma_f32_16x16x32_f16(
                false, a, false, cb, (short)0, acc[s], false, false);
            if (s < 3) cb = nb;
        }
        __syncthreads();
    }

    const int mbase = m0 + wm * 16 + ((lane >> 4) << 3);
    const int nbase = n0 + wn * 64 + (lane & 15);
#pragma unroll
    for (int s = 0; s < 4; ++s) {
        const int n = nbase + s * 16;
        const float bias = b2[(size_t)e * D_IN + n];
#pragma unroll
        for (int i = 0; i < 8; ++i) {
            const int row = mbase + i;
            if (row < count) {
                const int tok = toklist[e * CAP + row];
                out[(size_t)tok * D_IN + n] = acc[s][i] + bias;
            }
        }
    }
}

// ---------------------------------------------------------------------------
// Launch
// ---------------------------------------------------------------------------
extern "C" void kernel_launch(void* const* d_in, const int* in_sizes, int n_in,
                              void* d_out, int out_size, void* d_ws, size_t ws_size,
                              hipStream_t stream) {
    (void)in_sizes; (void)n_in; (void)out_size; (void)ws_size;

    const float* x   = (const float*)d_in[0];
    const float* W1  = (const float*)d_in[1];
    const float* b1  = (const float*)d_in[2];
    const float* W2  = (const float*)d_in[3];
    const float* b2  = (const float*)d_in[4];
    const float* W3  = (const float*)d_in[5];
    const float* b3  = (const float*)d_in[6];
    const int*   idx = (const int*)d_in[7];
    float* out = (float*)d_out;

    char* ws = (char*)d_ws;
    int* counts  = (int*)ws;                 // 32 ints
    int* toklist = (int*)(ws + 256);         // 32*512 ints (64 KB)
    _Float16* H  = (_Float16*)(ws + 131072); // 32*512*1024 f16 (32 MB)

    moe_dispatch_kernel<<<NEXP, 32, 0, stream>>>(idx, counts, toklist);
    moe_ffn1_kernel<<<dim3(CAP / 32, D_INTER / 256, NEXP), 256, 0, stream>>>(
        x, W1, b1, W3, b3, counts, toklist, H);
    moe_ffn2_kernel<<<dim3(CAP / 32, D_IN / 256, NEXP), 256, 0, stream>>>(
        H, W2, b2, counts, toklist, out);
}
